// GATStockPredictionModel_22247930593605
// MI455X (gfx1250) — matrix operand order repaired and checked by reading, not compiled, not run on hardware
//
#include <hip/hip_runtime.h>
#include <hip/hip_bf16.h>
#include <stdint.h>

// GAT forward, MI455X (gfx1250, wave32, WMMA).
// Pipeline:
//   1. convert x -> bf16 (row-major), W -> bf16 transposed (feature-major)
//   2. k_project: H = X@W via v_wmma_f32_16x16x32_bf16; emits f1/f2 = H@a halves,
//      and H^T in bf16 (feature-major) so aggregation B-fragments are contiguous.
//   3. k_stats: per-row online softmax max/sum over masked leakyrelu(f1_i+f2_j),
//      wave32 shuffle-xor reduction. exp(-9e15 - m) == 0 in f32, so skipping
//      masked entries matches the reference exactly.
//   4. k_aggregate: fused softmax @ H. Builds 16x32 bf16 P tile in registers in
//      the ISA A-matrix lane layout, WMMAs against H^T bf16 B tiles (two 16B loads),
//      scales rows by 1/l, applies elu (layer 1) -> hcat bf16 / out f32.
// All heavy math goes through v_wmma_f32_16x16x32_bf16 (f32 accumulate).

#define Nn    4096
#define FEAT  64
#define NH    4
#define HF    256   // NH * FEAT

typedef __bf16 bf16_t;
typedef __attribute__((ext_vector_type(16))) __bf16 v16bf;
typedef __attribute__((ext_vector_type(8)))  __bf16 v8bf;
typedef __attribute__((ext_vector_type(8)))  float  v8f;

__device__ __forceinline__ float lrelu(float x) { return x > 0.f ? x : 0.2f * x; }

// ---------------------------------------------------------------- conversions
__global__ __launch_bounds__(256) void k_f32_to_bf16(const float* __restrict__ src,
                                                     bf16_t* __restrict__ dst, int n) {
  int i = blockIdx.x * blockDim.x + threadIdx.x;
  if (i < n) dst[i] = (bf16_t)src[i];
}

// src batch b: [K][Nc] f32 -> dst batch b: [Nc][K] bf16
__global__ __launch_bounds__(256) void k_transpose_bf16(const float* __restrict__ src,
                                                        bf16_t* __restrict__ dst,
                                                        int K, int Nc) {
  int b = blockIdx.y;
  int idx = blockIdx.x * blockDim.x + threadIdx.x;
  if (idx >= K * Nc) return;
  int k = idx / Nc, n = idx % Nc;
  size_t base = (size_t)b * K * Nc;
  dst[base + (size_t)n * K + k] = (bf16_t)src[base + (size_t)k * Nc + n];
}

// ---------------------------------------------------------------- projection
// One wave per 16-row tile per head. H-tile = Abf(16xKin) @ W(Kinx64), bf16 WMMA.
__global__ __launch_bounds__(32) void k_project(const bf16_t* __restrict__ Abf,  // [Nn][Kin]
                                                const bf16_t* __restrict__ WT,   // [head][64][Kin]
                                                const float*  __restrict__ avec, // [head][128]
                                                bf16_t* __restrict__ HT,         // [head][64][Nn]
                                                float* __restrict__ f1,          // [head][Nn]
                                                float* __restrict__ f2,          // [head][Nn]
                                                int Kin) {
  __shared__ float hs[16][FEAT + 1];   // +1: break 64-bank conflicts on row reads
  const int head = blockIdx.y;
  const int i0   = blockIdx.x * 16;
  const int l    = threadIdx.x;        // 0..31, one wave
  const int mr   = l & 15;             // A-row / B-col for this lane
  const int koff = (l < 16) ? 0 : 8;   // A-matrix K sub-offset (ISA 16-bit A layout)
  const int khalf= (l < 16) ? 0 : 16;  // unused; B loads use koff, same K order as A

  const bf16_t* Arow = Abf + (size_t)(i0 + mr) * Kin;
  const bf16_t* WTh  = WT + (size_t)head * FEAT * Kin;

  v8f acc[4];
  #pragma unroll
  for (int t = 0; t < 4; ++t)
    #pragma unroll
    for (int e = 0; e < 8; ++e) acc[t][e] = 0.f;

  for (int k0 = 0; k0 < Kin; k0 += 32) {
    v8bf alo = *(const v8bf*)(Arow + k0 + koff);        // K = koff..koff+7
    v8bf ahi = *(const v8bf*)(Arow + k0 + 16 + koff);   // K = 16+koff..
    v16bf a;
    #pragma unroll
    for (int e = 0; e < 8; ++e) { a[e] = alo[e]; a[8 + e] = ahi[e]; }
    #pragma unroll
    for (int t = 0; t < 4; ++t) {
      const bf16_t* Wp = WTh + (size_t)(t * 16 + mr) * Kin + k0;
      v8bf blo = *(const v8bf*)(Wp + koff);
      v8bf bhi = *(const v8bf*)(Wp + 16 + koff);
      v16bf b;
      #pragma unroll
      for (int e = 0; e < 8; ++e) { b[e] = blo[e]; b[8 + e] = bhi[e]; }
      acc[t] = __builtin_amdgcn_wmma_f32_16x16x32_bf16(false, a, false, b,
                                                       (short)0, acc[t], false, false);
    }
  }

  // Spill f32 tile to LDS (C/D layout: lane<16 -> rows 0..7, lane>=16 -> rows 8..15)
  const int rhi = (l < 16) ? 0 : 8;
  #pragma unroll
  for (int t = 0; t < 4; ++t)
    #pragma unroll
    for (int v = 0; v < 8; ++v) hs[rhi + v][t * 16 + mr] = acc[t][v];
  __syncthreads();

  // f1/f2: lanes 0-15 -> f1 rows, lanes 16-31 -> f2 rows
  {
    const float* av = avec + (size_t)head * 2 * FEAT + ((l < 16) ? 0 : FEAT);
    float s = 0.f;
    for (int c = 0; c < FEAT; ++c) s += hs[mr][c] * av[c];
    if (l < 16) f1[(size_t)head * Nn + i0 + mr] = s;
    else        f2[(size_t)head * Nn + i0 + mr] = s;
  }

  // H^T bf16 (feature-major): 16 contiguous bf16 per column per tile
  bf16_t* HTh = HT + (size_t)head * FEAT * Nn;
  for (int c = l; c < FEAT; c += 32)
    for (int r = 0; r < 16; ++r)
      HTh[(size_t)c * Nn + i0 + r] = (bf16_t)hs[r][c];
}

// ---------------------------------------------------------------- softmax stats
// One wave per row; all heads per adj read. Online max/sum, wave32 reduction.
template <int NHEADS>
__global__ __launch_bounds__(32) void k_stats(const uint8_t* __restrict__ adj,
                                              const float* __restrict__ f1,
                                              const float* __restrict__ f2,
                                              float* __restrict__ mrow,
                                              float* __restrict__ linv) {
  const int i = blockIdx.x;
  const int l = threadIdx.x;
  const uint8_t* arow = adj + (size_t)i * Nn;

  float m[NHEADS], s[NHEADS], f1i[NHEADS];
  #pragma unroll
  for (int h = 0; h < NHEADS; ++h) {
    m[h] = -3.0e38f; s[h] = 0.f; f1i[h] = f1[(size_t)h * Nn + i];
  }
  for (int j = l; j < Nn; j += 32) {
    if (arow[j]) {
      #pragma unroll
      for (int h = 0; h < NHEADS; ++h) {
        float e = lrelu(f1i[h] + f2[(size_t)h * Nn + j]);
        if (e > m[h]) { s[h] = s[h] * __expf(m[h] - e) + 1.f; m[h] = e; }
        else          { s[h] += __expf(e - m[h]); }
      }
    }
  }
  // wave32 combine of (max, scaled-sum)
  for (int off = 16; off >= 1; off >>= 1) {
    #pragma unroll
    for (int h = 0; h < NHEADS; ++h) {
      float m2 = __shfl_xor(m[h], off, 32);
      float s2 = __shfl_xor(s[h], off, 32);
      float M  = fmaxf(m[h], m2);
      s[h] = s[h] * __expf(m[h] - M) + s2 * __expf(m2 - M);
      m[h] = M;
    }
  }
  if (l == 0) {
    #pragma unroll
    for (int h = 0; h < NHEADS; ++h) {
      mrow[(size_t)h * Nn + i] = m[h];
      linv[(size_t)h * Nn + i] = 1.f / s[h];
    }
  }
}

// ---------------------------------------------------------------- fused softmax @ H
// One wave per 16-row tile per head. P(16x32) built in registers in the ISA
// bf16 A-matrix layout; V fragments are two 16B loads per lane from H^T.
template <bool IS_OUT>
__global__ __launch_bounds__(32) void k_aggregate(const uint8_t* __restrict__ adj,
                                                  const float* __restrict__ f1,
                                                  const float* __restrict__ f2,
                                                  const float* __restrict__ mrow,
                                                  const float* __restrict__ linv,
                                                  const bf16_t* __restrict__ HT, // [head][64][Nn]
                                                  bf16_t* __restrict__ hcat,     // layer1: [Nn][HF]
                                                  float*  __restrict__ outp) {   // layer2: [Nn][64]
  const int head  = blockIdx.y;
  const int i0    = blockIdx.x * 16;
  const int l     = threadIdx.x;
  const int mr    = l & 15;
  const int row   = i0 + mr;
  const int koff  = (l < 16) ? 0 : 8;
  const int khalf = (l < 16) ? 0 : 16;

  const uint8_t* arow = adj + (size_t)row * Nn;
  const float*   f2h  = f2 + (size_t)head * Nn;
  const bf16_t*  HTh  = HT + (size_t)head * FEAT * (size_t)Nn;
  const float    f1r  = f1[(size_t)head * Nn + row];
  const float    mR   = mrow[(size_t)head * Nn + row];

  v8f acc[4];
  #pragma unroll
  for (int t = 0; t < 4; ++t)
    #pragma unroll
    for (int e = 0; e < 8; ++e) acc[t][e] = 0.f;

  for (int j0 = 0; j0 < Nn; j0 += 32) {
    if (j0 + 1024 < Nn) __builtin_prefetch(arow + j0 + 1024, 0, 1);
    const uint64_t bits0 = *(const uint64_t*)(arow + j0 + koff);
    const uint64_t bits1 = *(const uint64_t*)(arow + j0 + 16 + koff);
    const v8f g0 = *(const v8f*)(f2h + j0 + koff);
    const v8f g1 = *(const v8f*)(f2h + j0 + 16 + koff);
    v16bf a;
    #pragma unroll
    for (int e = 0; e < 8; ++e) {
      float p0 = __expf(lrelu(f1r + g0[e]) - mR);
      float p1 = __expf(lrelu(f1r + g1[e]) - mR);
      p0 = ((bits0 >> (8 * e)) & 0xffull) ? p0 : 0.f;   // masked: exp(-9e15-m)==0
      p1 = ((bits1 >> (8 * e)) & 0xffull) ? p1 : 0.f;
      a[e]     = (bf16_t)p0;
      a[8 + e] = (bf16_t)p1;
    }
    #pragma unroll
    for (int t = 0; t < 4; ++t) {
      const bf16_t* Hp = HTh + (size_t)(t * 16 + mr) * Nn + j0;
      v8bf blo = *(const v8bf*)(Hp + koff);
      v8bf bhi = *(const v8bf*)(Hp + 16 + koff);
      v16bf b;
      #pragma unroll
      for (int e = 0; e < 8; ++e) { b[e] = blo[e]; b[8 + e] = bhi[e]; }
      acc[t] = __builtin_amdgcn_wmma_f32_16x16x32_bf16(false, a, false, b,
                                                       (short)0, acc[t], false, false);
    }
  }

  const int rhi = (l < 16) ? 0 : 8;
  #pragma unroll
  for (int v = 0; v < 8; ++v) {
    const int r = i0 + rhi + v;
    const float lv = linv[(size_t)head * Nn + r];
    #pragma unroll
    for (int t = 0; t < 4; ++t) {
      float val = acc[t][v] * lv;
      if (IS_OUT) {
        outp[(size_t)r * FEAT + t * 16 + mr] = val;               // layer 2: no elu
      } else {
        val = val > 0.f ? val : (__expf(val) - 1.f);              // elu
        hcat[(size_t)r * HF + head * FEAT + t * 16 + mr] = (bf16_t)val;
      }
    }
  }
}

// ---------------------------------------------------------------- host launch
extern "C" void kernel_launch(void* const* d_in, const int* in_sizes, int n_in,
                              void* d_out, int out_size, void* d_ws, size_t ws_size,
                              hipStream_t stream) {
  (void)in_sizes; (void)n_in; (void)out_size; (void)ws_size;
  const float*   x       = (const float*)d_in[0];
  const float*   W_heads = (const float*)d_in[1];
  const float*   a_heads = (const float*)d_in[2];
  const float*   W_out   = (const float*)d_in[3];
  const float*   a_out   = (const float*)d_in[4];
  const uint8_t* adj     = (const uint8_t*)d_in[5];  // jax bool: 1 byte/elem
  float* out = (float*)d_out;

  char* p = (char*)d_ws;
  auto alloc = [&](size_t bytes) -> char* {
    char* r = p; p += (bytes + 255) & ~(size_t)255; return r;
  };
  bf16_t* Xbf  = (bf16_t*)alloc((size_t)Nn * FEAT * 2);
  bf16_t* WT1  = (bf16_t*)alloc((size_t)NH * FEAT * FEAT * 2);
  bf16_t* WT2  = (bf16_t*)alloc((size_t)FEAT * HF * 2);
  bf16_t* HT1  = (bf16_t*)alloc((size_t)NH * FEAT * Nn * 2);
  bf16_t* HT2  = (bf16_t*)alloc((size_t)FEAT * Nn * 2);
  bf16_t* hcat = (bf16_t*)alloc((size_t)Nn * HF * 2);
  float* f11 = (float*)alloc((size_t)NH * Nn * 4);
  float* f21 = (float*)alloc((size_t)NH * Nn * 4);
  float* m1  = (float*)alloc((size_t)NH * Nn * 4);
  float* li1 = (float*)alloc((size_t)NH * Nn * 4);
  float* f12 = (float*)alloc((size_t)Nn * 4);
  float* f22 = (float*)alloc((size_t)Nn * 4);
  float* m2  = (float*)alloc((size_t)Nn * 4);
  float* li2 = (float*)alloc((size_t)Nn * 4);

  // --- conversions
  k_f32_to_bf16<<<(Nn * FEAT + 255) / 256, 256, 0, stream>>>(x, Xbf, Nn * FEAT);
  k_transpose_bf16<<<dim3((FEAT * FEAT + 255) / 256, NH), 256, 0, stream>>>(W_heads, WT1, FEAT, FEAT);
  k_transpose_bf16<<<dim3((HF * FEAT + 255) / 256, 1), 256, 0, stream>>>(W_out, WT2, HF, FEAT);

  // --- layer 1
  k_project<<<dim3(Nn / 16, NH), 32, 0, stream>>>(Xbf, WT1, a_heads, HT1, f11, f21, FEAT);
  k_stats<NH><<<Nn, 32, 0, stream>>>(adj, f11, f21, m1, li1);
  k_aggregate<false><<<dim3(Nn / 16, NH), 32, 0, stream>>>(adj, f11, f21, m1, li1, HT1, hcat, nullptr);

  // --- layer 2
  k_project<<<dim3(Nn / 16, 1), 32, 0, stream>>>(hcat, WT2, a_out, HT2, f12, f22, HF);
  k_stats<1><<<Nn, 32, 0, stream>>>(adj, f12, f22, m2, li2);
  k_aggregate<true><<<dim3(Nn / 16, 1), 32, 0, stream>>>(adj, f12, f22, m2, li2, HT2, nullptr, out);
}
